// AudioGraphEncoder_73624329388223
// MI455X (gfx1250) — matrix operand order repaired and checked by reading, not compiled, not run on hardware
//
#include <hip/hip_runtime.h>
#include <hip/hip_bf16.h>
#include <stdint.h>

// ---------------------------------------------------------------------------
// Problem constants (from reference)
// ---------------------------------------------------------------------------
#define NN 8192
#define DD 1024
#define HH 256
#define CC 7
#define KK 8
#define TWc 1.0f
#define SROWS 32   // rows per block in sim kernel

typedef __bf16 bf16;
typedef __attribute__((ext_vector_type(16))) __bf16 v16bf;
typedef __attribute__((ext_vector_type(8)))  float  v8f;

// ---------------------------------------------------------------------------
// WMMA helpers (gfx1250, wave32).  16x16x32 bf16 -> f32 accumulate.
// A fragment: row-major A[M x K], lane = 16*half + m.
//   a[j]     = A[m][kBase + 16*half + j]              (contiguous, 32B)
// B fragment from TRANSPOSED operand Bt[N x K] (so B[k][n] = Bt[n][k]):
//   b[j]     = Bt[nBase + n][kBase + 16*half + j]     (contiguous, 32B)
// C/D: lane = 16*half + n, c[r] = C[mBase + r + 8*half][nBase + n]
// ---------------------------------------------------------------------------
__device__ __forceinline__ v8f wmma_bf16(v16bf a, v16bf b, v8f c) {
  return __builtin_amdgcn_wmma_f32_16x16x32_bf16(false, a, false, b,
                                                 (short)0, c, false, false);
}

__device__ __forceinline__ v16bf load_a_frag(const bf16* A, int lda,
                                             int mBase, int kBase, int lane) {
  const int m = lane & 15, half = lane >> 4;
  const bf16* row = A + (size_t)(mBase + m) * lda + kBase + 16 * half;
  v16bf a;
#pragma unroll
  for (int j = 0; j < 8; ++j)  a[j] = row[j];
#pragma unroll
  for (int j = 8; j < 16; ++j) a[j] = row[j];
  return a;
}

__device__ __forceinline__ v16bf load_bt_frag(const bf16* Bt, int ldb,
                                              int kBase, int nBase, int lane) {
  const int n = lane & 15, half = lane >> 4;
  const bf16* row = Bt + (size_t)(nBase + n) * ldb + kBase + 16 * half;
  v16bf b;
#pragma unroll
  for (int j = 0; j < 16; ++j) b[j] = row[j];
  return b;
}

// per-thread sorted (desc) top-9 insertion
#define TOP9_INSERT(v, col)                                              \
  if ((v) > val9[8]) {                                                   \
    val9[8] = (v); idx9[8] = (col);                                      \
    _Pragma("unroll")                                                    \
    for (int p = 8; p > 0; --p) {                                        \
      if (val9[p] > val9[p - 1]) {                                       \
        float tv = val9[p]; val9[p] = val9[p - 1]; val9[p - 1] = tv;     \
        int   ti = idx9[p]; idx9[p] = idx9[p - 1]; idx9[p - 1] = ti;     \
      }                                                                  \
    }                                                                    \
  }

// ---------------------------------------------------------------------------
// 1. BatchNorm + row L2-norm:  h = BN(x); xn = h / (||h|| + 1e-8)
// ---------------------------------------------------------------------------
__global__ void k_bn_norm(const float* __restrict__ x,
                          const float* __restrict__ gamma,
                          const float* __restrict__ beta,
                          const float* __restrict__ mean,
                          const float* __restrict__ var,
                          float* __restrict__ h, bf16* __restrict__ hB,
                          bf16* __restrict__ xnB) {
  __shared__ float red[256];
  const int row = blockIdx.x, tid = threadIdx.x;
  float loc[4];
  float ss = 0.0f;
#pragma unroll
  for (int i = 0; i < 4; ++i) {
    const int c = i * 256 + tid;
    const float xv = x[(size_t)row * DD + c];
    const float hv = (xv - mean[c]) * rsqrtf(var[c] + 1e-5f) * gamma[c] + beta[c];
    loc[i] = hv;
    ss += hv * hv;
    h[(size_t)row * DD + c]  = hv;
    hB[(size_t)row * DD + c] = (bf16)hv;
  }
  red[tid] = ss;
  __syncthreads();
  for (int s = 128; s > 0; s >>= 1) {
    if (tid < s) red[tid] += red[tid + s];
    __syncthreads();
  }
  const float inv = 1.0f / (sqrtf(red[0]) + 1e-8f);
#pragma unroll
  for (int i = 0; i < 4; ++i) {
    const int c = i * 256 + tid;
    xnB[(size_t)row * DD + c] = (bf16)(loc[i] * inv);
  }
}

// ---------------------------------------------------------------------------
// 2. sim = xn @ xn^T, streamed top-9.
//    Block = 256 thr (8 waves) owns a 32-row strip.  Per 128-col chunk each
//    wave computes two 16x16 WMMA tiles (rows 0-15 / 16-31) sharing one B
//    fragment (double-buffered over K).  Tiles staged in LDS; the top-9 scan
//    is partitioned: 256 threads = 32 rows x 8 column-subsets, each keeping a
//    private sorted top-9; partials merged once at the end.
// ---------------------------------------------------------------------------
__global__ void __launch_bounds__(256)
k_sim_topk(const bf16* __restrict__ xn, int* __restrict__ nbr,
           float* __restrict__ wgt, float* __restrict__ mchain) {
  __shared__ float tile[SROWS][136];     // 128 cols + pad
  __shared__ float pv[SROWS][8][9];
  __shared__ int   pi[SROWS][8][9];

  const int tid = threadIdx.x;
  const int lane = tid & 31, wv = tid >> 5;
  const int mBase = blockIdx.x * SROWS;
  const int n_ = lane & 15, half = lane >> 4;
  const int srow = tid & 31;   // row this thread scans
  const int part = tid >> 5;   // column subset (mod 8)

  float val9[9];
  int   idx9[9];
#pragma unroll
  for (int i = 0; i < 9; ++i) { val9[i] = -1e30f; idx9[i] = 0; }

  for (int chunk = 0; chunk < NN; chunk += 128) {
    const int nBase = chunk + wv * 16;
    v8f acc0 = {}, acc1 = {};

    // software-pipelined K loop: B double-buffered, shared by both M-subtiles
    v16bf b = load_bt_frag(xn, DD, 0, nBase, lane);
    int k = 0;
    for (; k + 32 < DD; k += 32) {
      v16bf bn = load_bt_frag(xn, DD, k + 32, nBase, lane);
      v16bf a0 = load_a_frag(xn, DD, mBase, k, lane);
      v16bf a1 = load_a_frag(xn, DD, mBase + 16, k, lane);
      acc0 = wmma_bf16(a0, b, acc0);
      acc1 = wmma_bf16(a1, b, acc1);
      b = bn;
    }
    {
      v16bf a0 = load_a_frag(xn, DD, mBase, k, lane);
      v16bf a1 = load_a_frag(xn, DD, mBase + 16, k, lane);
      acc0 = wmma_bf16(a0, b, acc0);
      acc1 = wmma_bf16(a1, b, acc1);
    }

#pragma unroll
    for (int r = 0; r < 8; ++r) {
      tile[r + 8 * half][wv * 16 + n_]      = acc0[r];
      tile[16 + r + 8 * half][wv * 16 + n_] = acc1[r];
    }
    __syncthreads();

    // parallel scan: each thread covers cols j = part + 8*i of its row
#pragma unroll
    for (int i = 0; i < 16; ++i) {
      const int j = part + 8 * i;
      const float v = tile[srow][j];
      const int col = chunk + j;
      TOP9_INSERT(v, col)
    }
    __syncthreads();
  }

  // publish partial top-9 lists
#pragma unroll
  for (int q = 0; q < 9; ++q) {
    pv[srow][part][q] = val9[q];
    pi[srow][part][q] = idx9[q];
  }
  __syncthreads();

  if (tid < SROWS) {
    const int rowg = mBase + tid;
#pragma unroll
    for (int i = 0; i < 9; ++i) { val9[i] = -1e30f; idx9[i] = 0; }
    for (int pgrp = 0; pgrp < 8; ++pgrp) {
#pragma unroll
      for (int q = 0; q < 9; ++q) {
        const float v = pv[tid][pgrp][q];
        const int col = pi[tid][pgrp][q];
        TOP9_INSERT(v, col)
      }
    }
    int pres = 0;
#pragma unroll
    for (int k = 0; k < KK; ++k) {
      const int col = idx9[k + 1];       // slot 0 == self (sim = 1)
      float v = val9[k + 1];
      int d = col - rowg; if (d < 0) d = -d;
      if (d == 1) v += TWc;
      nbr[rowg * KK + k] = col;
      wgt[rowg * KK + k] = v;
      if (col == rowg + 1) pres = 1;
    }
    mchain[rowg] = pres ? 0.0f : TWc;
  }
}

// ---------------------------------------------------------------------------
// 3. Dual-input WMMA GEMM:  C[M,N] = A0@B0 (+ A1@B1) + bias
//    A: [M,K] bf16 row-major; B given TRANSPOSED as Bt[N,K] bf16.
//    Block 256 thr (8 waves): 128 rows x 32 cols; each wave 16 rows x 32 cols
//    (2 accumulators).  All fragments double-buffered over K.
// ---------------------------------------------------------------------------
__global__ void __launch_bounds__(256)
k_gemm2(const bf16* __restrict__ A0, const bf16* __restrict__ B0t,
        const bf16* __restrict__ A1, const bf16* __restrict__ B1t,
        const float* __restrict__ bias, float* __restrict__ C,
        int M, int Kd, int Nn) {
  const int lane = threadIdx.x & 31, wv = threadIdx.x >> 5;
  const int mBase = blockIdx.y * 128 + wv * 16;
  const int nBase = blockIdx.x * 32;
  (void)M;

  v8f acc0 = {}, acc1 = {};

  v16bf a0  = load_a_frag(A0, Kd, mBase, 0, lane);
  v16bf b00 = load_bt_frag(B0t, Kd, 0, nBase, lane);
  v16bf b01 = load_bt_frag(B0t, Kd, 0, nBase + 16, lane);
  v16bf a1, b10, b11;
  if (A1) {
    a1  = load_a_frag(A1, Kd, mBase, 0, lane);
    b10 = load_bt_frag(B1t, Kd, 0, nBase, lane);
    b11 = load_bt_frag(B1t, Kd, 0, nBase + 16, lane);
  }

  int k = 0;
  for (; k + 32 < Kd; k += 32) {
    const int kn = k + 32;
    v16bf a0n  = load_a_frag(A0, Kd, mBase, kn, lane);
    v16bf b00n = load_bt_frag(B0t, Kd, kn, nBase, lane);
    v16bf b01n = load_bt_frag(B0t, Kd, kn, nBase + 16, lane);
    acc0 = wmma_bf16(a0, b00, acc0);
    acc1 = wmma_bf16(a0, b01, acc1);
    if (A1) {  // uniform scalar branch: EXEC stays all-ones for WMMA
      v16bf a1n  = load_a_frag(A1, Kd, mBase, kn, lane);
      v16bf b10n = load_bt_frag(B1t, Kd, kn, nBase, lane);
      v16bf b11n = load_bt_frag(B1t, Kd, kn, nBase + 16, lane);
      acc0 = wmma_bf16(a1, b10, acc0);
      acc1 = wmma_bf16(a1, b11, acc1);
      a1 = a1n; b10 = b10n; b11 = b11n;
    }
    a0 = a0n; b00 = b00n; b01 = b01n;
  }
  acc0 = wmma_bf16(a0, b00, acc0);
  acc1 = wmma_bf16(a0, b01, acc1);
  if (A1) {
    acc0 = wmma_bf16(a1, b10, acc0);
    acc1 = wmma_bf16(a1, b11, acc1);
  }

  const int n_ = lane & 15, half = lane >> 4;
#pragma unroll
  for (int r = 0; r < 8; ++r) {
    const int m = mBase + r + 8 * half;
    C[(size_t)m * Nn + nBase + n_]      = acc0[r] + bias[nBase + n_];
    C[(size_t)m * Nn + nBase + 16 + n_] = acc1[r] + bias[nBase + 16 + n_];
  }
}

// ---------------------------------------------------------------------------
// 4. Aggregation (scatter-add) kernels
// ---------------------------------------------------------------------------
__global__ void k_zero_f32(float* __restrict__ p, long long n) {
  long long i = (long long)blockIdx.x * blockDim.x + threadIdx.x;
  const long long stride = (long long)gridDim.x * blockDim.x;
  for (; i < n; i += stride) p[i] = 0.0f;
}

__global__ void k_agg_edges(const float* __restrict__ h, const int* __restrict__ nbr,
                            const float* __restrict__ wgt, float* __restrict__ agg,
                            int F) {
  const long long idx = (long long)blockIdx.x * blockDim.x + threadIdx.x;
  const long long total = (long long)NN * KK * F;
  if (idx >= total) return;
  const int f = (int)(idx % F);
  const long long e = idx / F;
  const int src = (int)(e >> 3);
  const int k   = (int)(e & 7);
  const int tgt = nbr[src * KK + k];
  const float wv = wgt[src * KK + k];
  atomicAdd(&agg[(size_t)tgt * F + f], h[(size_t)src * F + f] * wv);
}

__global__ void k_agg_chain(const float* __restrict__ h, const float* __restrict__ mchain,
                            float* __restrict__ agg, int F) {
  const long long idx = (long long)blockIdx.x * blockDim.x + threadIdx.x;
  const long long total = (long long)(NN - 1) * F;
  if (idx >= total) return;
  const int f = (int)(idx % F);
  const int i = (int)(idx / F);
  const float m = mchain[i];
  if (m != 0.0f) {
    atomicAdd(&agg[(size_t)(i + 1) * F + f], m * h[(size_t)i * F + f]);
    atomicAdd(&agg[(size_t)i * F + f],       m * h[(size_t)(i + 1) * F + f]);
  }
}

// ---------------------------------------------------------------------------
// 5. Converters
// ---------------------------------------------------------------------------
__global__ void k_f32_to_bf16(const float* __restrict__ in, bf16* __restrict__ out,
                              long long n) {
  long long i = (long long)blockIdx.x * blockDim.x + threadIdx.x;
  const long long stride = (long long)gridDim.x * blockDim.x;
  for (; i < n; i += stride) out[i] = (bf16)in[i];
}

// in: [Kd][Nn] f32 row-major  ->  out: [Nn][Kd] bf16 (transposed)
__global__ void k_transpose_bf16(const float* __restrict__ in, bf16* __restrict__ out,
                                 int Kd, int Nn) {
  const long long idx = (long long)blockIdx.x * blockDim.x + threadIdx.x;
  if (idx >= (long long)Kd * Nn) return;
  const int k = (int)(idx / Nn);
  const int n = (int)(idx % Nn);
  out[(size_t)n * Kd + k] = (bf16)in[(size_t)k * Nn + n];
}

// ---------------------------------------------------------------------------
// 6. relu + residual + LayerNorm epilogue (H = 256, one block per row)
// ---------------------------------------------------------------------------
__global__ void k_relu_res_ln(const float* __restrict__ y, const float* __restrict__ resid,
                              const float* __restrict__ g, const float* __restrict__ b,
                              float* __restrict__ out, bf16* __restrict__ outB) {
  __shared__ float s1[256], s2[256];
  const int row = blockIdx.x, tid = threadIdx.x;
  float t = y[(size_t)row * HH + tid];
  t = t > 0.0f ? t : 0.0f;
  t += resid[(size_t)row * HH + tid];
  s1[tid] = t;
  s2[tid] = t * t;
  __syncthreads();
  for (int s = 128; s > 0; s >>= 1) {
    if (tid < s) { s1[tid] += s1[tid + s]; s2[tid] += s2[tid + s]; }
    __syncthreads();
  }
  const float mean = s1[0] * (1.0f / HH);
  const float var  = s2[0] * (1.0f / HH) - mean * mean;
  const float o = (t - mean) * rsqrtf(var + 1e-5f) * g[tid] + b[tid];
  out[(size_t)row * HH + tid]  = o;
  outB[(size_t)row * HH + tid] = (bf16)o;
}

// ---------------------------------------------------------------------------
// 7. Final FC:  out[N,7] = h3 @ fc_W + fc_b   (tiny, VALU)
// ---------------------------------------------------------------------------
__global__ void k_fc(const float* __restrict__ h, const float* __restrict__ W,
                     const float* __restrict__ bias, float* __restrict__ out) {
  const int idx = blockIdx.x * blockDim.x + threadIdx.x;
  if (idx >= NN * CC) return;
  const int row = idx / CC, c = idx % CC;
  float s = bias[c];
#pragma unroll 8
  for (int k = 0; k < HH; ++k) s += h[(size_t)row * HH + k] * W[k * CC + c];
  out[idx] = s;
}

// ---------------------------------------------------------------------------
// Host orchestration
// ---------------------------------------------------------------------------
extern "C" void kernel_launch(void* const* d_in, const int* in_sizes, int n_in,
                              void* d_out, int out_size, void* d_ws, size_t ws_size,
                              hipStream_t stream) {
  (void)in_sizes; (void)n_in; (void)out_size; (void)ws_size;

  const float* x        = (const float*)d_in[0];
  const float* bn_gamma = (const float*)d_in[1];
  const float* bn_beta  = (const float*)d_in[2];
  const float* bn_mean  = (const float*)d_in[3];
  const float* bn_var   = (const float*)d_in[4];
  const float* res_W    = (const float*)d_in[5];
  const float* res_b    = (const float*)d_in[6];
  const float* c1_rel_W = (const float*)d_in[7];
  const float* c1_rel_b = (const float*)d_in[8];
  const float* c1_root_W= (const float*)d_in[9];
  const float* c2_rel_W = (const float*)d_in[10];
  const float* c2_rel_b = (const float*)d_in[11];
  const float* c2_root_W= (const float*)d_in[12];
  const float* c3_rel_W = (const float*)d_in[13];
  const float* c3_rel_b = (const float*)d_in[14];
  const float* c3_root_W= (const float*)d_in[15];
  const float* ln1_g    = (const float*)d_in[16];
  const float* ln1_b    = (const float*)d_in[17];
  const float* ln2_g    = (const float*)d_in[18];
  const float* ln2_b    = (const float*)d_in[19];
  const float* ln3_g    = (const float*)d_in[20];
  const float* ln3_b    = (const float*)d_in[21];
  const float* fc_W     = (const float*)d_in[22];
  const float* fc_b     = (const float*)d_in[23];
  float* out = (float*)d_out;

  // ---- carve workspace ----
  char* p = (char*)d_ws;
  auto carve = [&](size_t bytes) -> void* {
    void* q = (void*)p;
    p += (bytes + 255) & ~(size_t)255;
    return q;
  };
  float* hD     = (float*)carve((size_t)NN * DD * 4);   // BN output, fp32
  bf16*  hDB    = (bf16* )carve((size_t)NN * DD * 2);
  bf16*  xnB    = (bf16* )carve((size_t)NN * DD * 2);
  float* aggD   = (float*)carve((size_t)NN * DD * 4);
  bf16*  aggDB  = (bf16* )carve((size_t)NN * DD * 2);
  bf16*  wresT  = (bf16* )carve((size_t)DD * HH * 2);
  bf16*  w1relT = (bf16* )carve((size_t)DD * HH * 2);
  bf16*  w1rootT= (bf16* )carve((size_t)DD * HH * 2);
  bf16*  w2relT = (bf16* )carve((size_t)HH * HH * 2);
  bf16*  w2rootT= (bf16* )carve((size_t)HH * HH * 2);
  bf16*  w3relT = (bf16* )carve((size_t)HH * HH * 2);
  bf16*  w3rootT= (bf16* )carve((size_t)HH * HH * 2);
  int*   nbr    = (int*  )carve((size_t)NN * KK * 4);
  float* wgt    = (float*)carve((size_t)NN * KK * 4);
  float* mchain = (float*)carve((size_t)NN * 4);
  float* r      = (float*)carve((size_t)NN * HH * 4);
  float* y      = (float*)carve((size_t)NN * HH * 4);
  float* hA     = (float*)carve((size_t)NN * HH * 4);
  bf16*  hAB    = (bf16* )carve((size_t)NN * HH * 2);
  float* hB     = (float*)carve((size_t)NN * HH * 4);
  bf16*  hBB    = (bf16* )carve((size_t)NN * HH * 2);
  float* aggH   = (float*)carve((size_t)NN * HH * 4);
  bf16*  aggHB  = (bf16* )carve((size_t)NN * HH * 2);

  const dim3 blk(256);

  // ---- weight transposes (f32 [K][N] -> bf16 [N][K]) ----
  k_transpose_bf16<<<(DD * HH + 255) / 256, blk, 0, stream>>>(res_W,     wresT,   DD, HH);
  k_transpose_bf16<<<(DD * HH + 255) / 256, blk, 0, stream>>>(c1_rel_W,  w1relT,  DD, HH);
  k_transpose_bf16<<<(DD * HH + 255) / 256, blk, 0, stream>>>(c1_root_W, w1rootT, DD, HH);
  k_transpose_bf16<<<(HH * HH + 255) / 256, blk, 0, stream>>>(c2_rel_W,  w2relT,  HH, HH);
  k_transpose_bf16<<<(HH * HH + 255) / 256, blk, 0, stream>>>(c2_root_W, w2rootT, HH, HH);
  k_transpose_bf16<<<(HH * HH + 255) / 256, blk, 0, stream>>>(c3_rel_W,  w3relT,  HH, HH);
  k_transpose_bf16<<<(HH * HH + 255) / 256, blk, 0, stream>>>(c3_root_W, w3rootT, HH, HH);

  // ---- BN + row norm ----
  k_bn_norm<<<NN, blk, 0, stream>>>(x, bn_gamma, bn_beta, bn_mean, bn_var, hD, hDB, xnB);

  // ---- sim + top-k graph build (WMMA-heavy) ----
  k_sim_topk<<<NN / SROWS, blk, 0, stream>>>(xnB, nbr, wgt, mchain);

  // ---- residual projection: r = h @ res_W + res_b ----
  k_gemm2<<<dim3(HH / 32, NN / 128), blk, 0, stream>>>(
      hDB, wresT, (const bf16*)nullptr, (const bf16*)nullptr, res_b, r, NN, DD, HH);

  // ---- conv1: agg over D=1024, then y = agg@relW + h@rootW + b ----
  k_zero_f32<<<4096, blk, 0, stream>>>(aggD, (long long)NN * DD);
  k_agg_edges<<<(int)(((long long)NN * KK * DD + 255) / 256), blk, 0, stream>>>(hD, nbr, wgt, aggD, DD);
  k_agg_chain<<<(int)(((long long)(NN - 1) * DD + 255) / 256), blk, 0, stream>>>(hD, mchain, aggD, DD);
  k_f32_to_bf16<<<4096, blk, 0, stream>>>(aggD, aggDB, (long long)NN * DD);
  k_gemm2<<<dim3(HH / 32, NN / 128), blk, 0, stream>>>(
      aggDB, w1relT, hDB, w1rootT, c1_rel_b, y, NN, DD, HH);
  k_relu_res_ln<<<NN, blk, 0, stream>>>(y, r, ln1_g, ln1_b, hA, hAB);

  // ---- conv2 ----
  k_zero_f32<<<2048, blk, 0, stream>>>(aggH, (long long)NN * HH);
  k_agg_edges<<<(int)(((long long)NN * KK * HH + 255) / 256), blk, 0, stream>>>(hA, nbr, wgt, aggH, HH);
  k_agg_chain<<<(int)(((long long)(NN - 1) * HH + 255) / 256), blk, 0, stream>>>(hA, mchain, aggH, HH);
  k_f32_to_bf16<<<2048, blk, 0, stream>>>(aggH, aggHB, (long long)NN * HH);
  k_gemm2<<<dim3(HH / 32, NN / 128), blk, 0, stream>>>(
      aggHB, w2relT, hAB, w2rootT, c2_rel_b, y, NN, HH, HH);
  k_relu_res_ln<<<NN, blk, 0, stream>>>(y, hA, ln2_g, ln2_b, hB, hBB);

  // ---- conv3 ----
  k_zero_f32<<<2048, blk, 0, stream>>>(aggH, (long long)NN * HH);
  k_agg_edges<<<(int)(((long long)NN * KK * HH + 255) / 256), blk, 0, stream>>>(hB, nbr, wgt, aggH, HH);
  k_agg_chain<<<(int)(((long long)(NN - 1) * HH + 255) / 256), blk, 0, stream>>>(hB, mchain, aggH, HH);
  k_f32_to_bf16<<<2048, blk, 0, stream>>>(aggH, aggHB, (long long)NN * HH);
  k_gemm2<<<dim3(HH / 32, NN / 128), blk, 0, stream>>>(
      aggHB, w3relT, hBB, w3rootT, c3_rel_b, y, NN, HH, HH);
  k_relu_res_ln<<<NN, blk, 0, stream>>>(y, hB, ln3_g, ln3_b, hA, hAB);

  // ---- final FC ----
  k_fc<<<(NN * CC + 255) / 256, blk, 0, stream>>>(hA, fc_W, fc_b, out);
}
